// TransformerBlock_82248623718843
// MI455X (gfx1250) — compile-verified, matrix operand order checked by reading, not run on hardware
//
#include <hip/hip_runtime.h>
#include <hip/hip_bf16.h>
#include <math.h>

typedef __attribute__((ext_vector_type(16))) __bf16 v16bf;
typedef __attribute__((ext_vector_type(8)))  float  v8f;

constexpr int B_ = 2, S_ = 1024, D_ = 1024, H_ = 16, DK_ = 64, DFF_ = 4096;
constexpr int GS_ = 4, G_ = 256;
constexpr float EPS_ = 1e-5f;

// ---------- bf16 helpers (round-to-nearest-even, hi/lo split) ----------
__device__ __forceinline__ unsigned short bf16_rne(float f) {
  unsigned int u = __builtin_bit_cast(unsigned int, f);
  unsigned int r = u + 0x7fffu + ((u >> 16) & 1u);
  return (unsigned short)(r >> 16);
}
__device__ __forceinline__ float bf16_up(unsigned short h) {
  unsigned int u = ((unsigned int)h) << 16;
  return __builtin_bit_cast(float, u);
}
__device__ __forceinline__ void pack_hl(float f, unsigned short* ph, unsigned short* pl) {
  unsigned short h = bf16_rne(f);
  *ph = h;
  *pl = bf16_rne(f - bf16_up(h));
}

// A/B fragment per CDNA5 ISA 16-bit 16x32 layout:
// lane<16: VGPR v holds K = 2v (v<4) or 2v+8 (v>=4); lane>=16: +8.
__device__ __forceinline__ v16bf frag_ld(const unsigned short* p, int stride, int half) {
  v16bf f;
#pragma unroll
  for (int i = 0; i < 8; ++i) {
    int k = ((i < 4) ? (2 * i) : (2 * i + 8)) + 8 * half;
    f[2 * i]     = __builtin_bit_cast(__bf16, p[k * stride]);
    f[2 * i + 1] = __builtin_bit_cast(__bf16, p[(k + 1) * stride]);
  }
  return f;
}
__device__ __forceinline__ v8f wmma_bf16(v16bf a, v16bf b, v8f c) {
  return __builtin_amdgcn_wmma_f32_16x16x32_bf16(false, a, false, b, (short)0, c, false, false);
}
__device__ __forceinline__ float red_max16(float x) {
#pragma unroll
  for (int m = 1; m < 16; m <<= 1) x = fmaxf(x, __shfl_xor(x, m, 32));
  return x;
}
__device__ __forceinline__ float red_sum16(float x) {
#pragma unroll
  for (int m = 1; m < 16; m <<= 1) x += __shfl_xor(x, m, 32);
  return x;
}

// ---------- pointwise kernels ----------
__global__ void pack_f32_bf16hl(const float* __restrict__ in, unsigned short* __restrict__ oh,
                                unsigned short* __restrict__ ol, int n) {
  int i = blockIdx.x * blockDim.x + threadIdx.x;
  if (i < n) pack_hl(in[i], &oh[i], &ol[i]);
}

__global__ __launch_bounds__(256) void rmsnorm_pack(const float* __restrict__ x,
                                                    const float* __restrict__ w,
                                                    unsigned short* __restrict__ oh,
                                                    unsigned short* __restrict__ ol) {
  __shared__ float red[256];
  const int row = blockIdx.x, tid = threadIdx.x;
  const float* xr = &x[(size_t)row * D_];
  float ss = 0.f;
  for (int i = tid; i < D_; i += 256) { float v = xr[i]; ss += v * v; }
  red[tid] = ss; __syncthreads();
  for (int s = 128; s > 0; s >>= 1) { if (tid < s) red[tid] += red[tid + s]; __syncthreads(); }
  float rinv = rsqrtf(red[0] * (1.0f / D_) + EPS_);
  for (int i = tid; i < D_; i += 256) {
    float v = xr[i] * w[i] * rinv;
    pack_hl(v, &oh[(size_t)row * D_ + i], &ol[(size_t)row * D_ + i]);
  }
}

__global__ void rope_pack(const float* __restrict__ QF, const float* __restrict__ KF,
                          unsigned short* __restrict__ QHo, unsigned short* __restrict__ QLo,
                          unsigned short* __restrict__ KHo, unsigned short* __restrict__ KLo) {
  int idx = blockIdx.x * blockDim.x + threadIdx.x;
  if (idx >= B_ * S_ * H_ * (DK_ / 2)) return;
  int i = idx & 31;
  int hh = (idx >> 5) & (H_ - 1);
  int s = (idx >> 9) & (S_ - 1);
  int b = idx >> 19;
  size_t off = ((size_t)(b * S_ + s)) * D_ + hh * DK_ + 2 * i;
  float inv = __powf(10000.0f, -(float)(2 * i) / (float)DK_);
  float ang = (float)s * inv;
  float sn, c;
  __sincosf(ang, &sn, &c);
  float q1 = QF[off], q2 = QF[off + 1];
  pack_hl(q1 * c - q2 * sn, &QHo[off], &QLo[off]);
  pack_hl(q1 * sn + q2 * c, &QHo[off + 1], &QLo[off + 1]);
  float k1 = KF[off], k2 = KF[off + 1];
  pack_hl(k1 * c - k2 * sn, &KHo[off], &KLo[off]);
  pack_hl(k1 * sn + k2 * c, &KHo[off + 1], &KLo[off + 1]);
}

__global__ void swiglu_pack(const float* __restrict__ L1v, const float* __restrict__ L3v,
                            unsigned short* __restrict__ GHo, unsigned short* __restrict__ GLo, int n) {
  int i = blockIdx.x * blockDim.x + threadIdx.x;
  if (i < n) {
    float a = L1v[i];
    float si = a / (1.f + __expf(-a));
    pack_hl(si * L3v[i], &GHo[i], &GLo[i]);
  }
}

// ---------- GEMM: C[M,N] = A[M,K] * W[N,K]^T (+resid), bf16x3 via WMMA ----------
__global__ __launch_bounds__(256) void gemm_bf16x3(
    const unsigned short* __restrict__ Ah, const unsigned short* __restrict__ Al,
    const unsigned short* __restrict__ Wh, const unsigned short* __restrict__ Wl,
    const float* __restrict__ resid, float* __restrict__ C, int N, int K) {
  __shared__ unsigned short sAh[64 * 32], sAl[64 * 32], sWh[64 * 32], sWl[64 * 32];
  const int tid = threadIdx.x, wid = tid >> 5, lane = tid & 31;
  const int half = lane >> 4, mrow = lane & 15;
  const int m0 = 16 * (wid >> 1), n0 = 32 * (wid & 1);
  const int mBase = blockIdx.y * 64, nBase = blockIdx.x * 64;
  const int lrow = tid >> 2, lcol = (tid & 3) * 8;
  v8f c0 = {}, c1 = {};
  for (int kk = 0; kk < K; kk += 32) {
    const size_t ga = (size_t)(mBase + lrow) * K + kk + lcol;
    const size_t gw = (size_t)(nBase + lrow) * K + kk + lcol;
    *(uint4*)&sAh[lrow * 32 + lcol] = *(const uint4*)&Ah[ga];
    *(uint4*)&sAl[lrow * 32 + lcol] = *(const uint4*)&Al[ga];
    *(uint4*)&sWh[lrow * 32 + lcol] = *(const uint4*)&Wh[gw];
    *(uint4*)&sWl[lrow * 32 + lcol] = *(const uint4*)&Wl[gw];
    __syncthreads();
    if (kk + 32 < K) {  // pull next K panel toward L2/WGP$ (global_prefetch_b8)
      __builtin_prefetch(&Ah[ga + 32], 0, 1);
      __builtin_prefetch(&Wh[gw + 32], 0, 1);
    }
    v16bf ah  = frag_ld(&sAh[(m0 + mrow) * 32], 1, half);
    v16bf al  = frag_ld(&sAl[(m0 + mrow) * 32], 1, half);
    v16bf b0h = frag_ld(&sWh[(n0 + mrow) * 32], 1, half);
    v16bf b0l = frag_ld(&sWl[(n0 + mrow) * 32], 1, half);
    v16bf b1h = frag_ld(&sWh[(n0 + 16 + mrow) * 32], 1, half);
    v16bf b1l = frag_ld(&sWl[(n0 + 16 + mrow) * 32], 1, half);
    c0 = wmma_bf16(ah, b0h, c0); c0 = wmma_bf16(al, b0h, c0); c0 = wmma_bf16(ah, b0l, c0);
    c1 = wmma_bf16(ah, b1h, c1); c1 = wmma_bf16(al, b1h, c1); c1 = wmma_bf16(ah, b1l, c1);
    __syncthreads();
  }
#pragma unroll
  for (int r = 0; r < 8; ++r) {
    int row = mBase + m0 + r + 8 * half;
    int col = nBase + n0 + mrow;
    float v0 = c0[r], v1 = c1[r];
    if (resid) {
      v0 += resid[(size_t)row * N + col];
      v1 += resid[(size_t)row * N + col + 16];
    }
    C[(size_t)row * N + col] = v0;
    C[(size_t)row * N + col + 16] = v1;
  }
}

// ---------- Flash attention (causal), 4 waves x 16 q-rows per block ----------
__global__ __launch_bounds__(128) void attn_flash(
    const unsigned short* __restrict__ QH, const unsigned short* __restrict__ QL,
    const unsigned short* __restrict__ KH, const unsigned short* __restrict__ KL,
    const unsigned short* __restrict__ VH, const unsigned short* __restrict__ VL,
    float* __restrict__ AO) {
  __shared__ unsigned short sKh[32 * 64], sKl[32 * 64], sVh[32 * 64], sVl[32 * 64];
  __shared__ unsigned short sPh[4][16 * 32], sPl[4][16 * 32];
  const int b = blockIdx.z, h = blockIdx.y, q0 = blockIdx.x * 64;
  const int tid = threadIdx.x, wid = tid >> 5, lane = tid & 31;
  const int half = lane >> 4, mrow = lane & 15;
  const int qr = q0 + wid * 16;
  const size_t qoff = ((size_t)(b * S_ + qr + mrow)) * D_ + h * DK_;
  v16bf qh0 = frag_ld(&QH[qoff], 1, half), qh1 = frag_ld(&QH[qoff + 32], 1, half);
  v16bf ql0 = frag_ld(&QL[qoff], 1, half), ql1 = frag_ld(&QL[qoff + 32], 1, half);
  float mrun[8], lrun[8];
  v8f o0 = {}, o1 = {}, o2 = {}, o3 = {};
#pragma unroll
  for (int r = 0; r < 8; ++r) { mrun[r] = -1e30f; lrun[r] = 0.f; }
  const int lrow = tid >> 2, lcol = (tid & 3) * 16;
  const int nkt = (q0 >> 5) + 2;  // keys [0, q0+63]
  for (int kt = 0; kt < nkt; ++kt) {
    const int kb = kt * 32;
    const size_t g = ((size_t)(b * S_ + kb + lrow)) * D_ + h * DK_ + lcol;
    *(uint4*)&sKh[lrow * 64 + lcol]     = *(const uint4*)&KH[g];
    *(uint4*)&sKh[lrow * 64 + lcol + 8] = *(const uint4*)&KH[g + 8];
    *(uint4*)&sKl[lrow * 64 + lcol]     = *(const uint4*)&KL[g];
    *(uint4*)&sKl[lrow * 64 + lcol + 8] = *(const uint4*)&KL[g + 8];
    *(uint4*)&sVh[lrow * 64 + lcol]     = *(const uint4*)&VH[g];
    *(uint4*)&sVh[lrow * 64 + lcol + 8] = *(const uint4*)&VH[g + 8];
    *(uint4*)&sVl[lrow * 64 + lcol]     = *(const uint4*)&VL[g];
    *(uint4*)&sVl[lrow * 64 + lcol + 8] = *(const uint4*)&VL[g + 8];
    __syncthreads();
    // scores: S[16q x 32k] = Q(16x64) * K(32x64)^T, bf16x3 over two 32-wide d-steps
    v8f s0 = {}, s1 = {};
    {
      v16bf kb0h = frag_ld(&sKh[mrow * 64], 1, half);
      v16bf kb0l = frag_ld(&sKl[mrow * 64], 1, half);
      s0 = wmma_bf16(qh0, kb0h, s0); s0 = wmma_bf16(ql0, kb0h, s0); s0 = wmma_bf16(qh0, kb0l, s0);
      v16bf kb1h = frag_ld(&sKh[mrow * 64 + 32], 1, half);
      v16bf kb1l = frag_ld(&sKl[mrow * 64 + 32], 1, half);
      s0 = wmma_bf16(qh1, kb1h, s0); s0 = wmma_bf16(ql1, kb1h, s0); s0 = wmma_bf16(qh1, kb1l, s0);
    }
    {
      v16bf kb0h = frag_ld(&sKh[(16 + mrow) * 64], 1, half);
      v16bf kb0l = frag_ld(&sKl[(16 + mrow) * 64], 1, half);
      s1 = wmma_bf16(qh0, kb0h, s1); s1 = wmma_bf16(ql0, kb0h, s1); s1 = wmma_bf16(qh0, kb0l, s1);
      v16bf kb1h = frag_ld(&sKh[(16 + mrow) * 64 + 32], 1, half);
      v16bf kb1l = frag_ld(&sKl[(16 + mrow) * 64 + 32], 1, half);
      s1 = wmma_bf16(qh1, kb1h, s1); s1 = wmma_bf16(ql1, kb1h, s1); s1 = wmma_bf16(qh1, kb1l, s1);
    }
    // online softmax (per-row reductions across 16 lanes within each half)
#pragma unroll
    for (int r = 0; r < 8; ++r) {
      int rowg = qr + r + 8 * half;
      float v0 = s0[r] * 0.125f; if (kb + mrow > rowg) v0 = -1e30f;
      float v1 = s1[r] * 0.125f; if (kb + 16 + mrow > rowg) v1 = -1e30f;
      float rmax = red_max16(fmaxf(v0, v1));
      float mnew = fmaxf(mrun[r], rmax);
      float a = __expf(mrun[r] - mnew);
      float p0 = __expf(v0 - mnew), p1 = __expf(v1 - mnew);
      float rs = red_sum16(p0 + p1);
      lrun[r] = lrun[r] * a + rs;
      mrun[r] = mnew;
      int prow = r + 8 * half;
      pack_hl(p0, &sPh[wid][prow * 32 + mrow],      &sPl[wid][prow * 32 + mrow]);
      pack_hl(p1, &sPh[wid][prow * 32 + 16 + mrow], &sPl[wid][prow * 32 + 16 + mrow]);
      o0[r] *= a; o1[r] *= a; o2[r] *= a; o3[r] *= a;
    }
    __syncthreads();
    // O += P(16x32) * V(32x64): K=32 exactly one bf16 WMMA step, 4 N-tiles
    v16bf pah = frag_ld(&sPh[wid][mrow * 32], 1, half);
    v16bf pal = frag_ld(&sPl[wid][mrow * 32], 1, half);
    {
      v16bf vbh = frag_ld(&sVh[mrow], 64, half);
      v16bf vbl = frag_ld(&sVl[mrow], 64, half);
      o0 = wmma_bf16(pah, vbh, o0); o0 = wmma_bf16(pal, vbh, o0); o0 = wmma_bf16(pah, vbl, o0);
    }
    {
      v16bf vbh = frag_ld(&sVh[16 + mrow], 64, half);
      v16bf vbl = frag_ld(&sVl[16 + mrow], 64, half);
      o1 = wmma_bf16(pah, vbh, o1); o1 = wmma_bf16(pal, vbh, o1); o1 = wmma_bf16(pah, vbl, o1);
    }
    {
      v16bf vbh = frag_ld(&sVh[32 + mrow], 64, half);
      v16bf vbl = frag_ld(&sVl[32 + mrow], 64, half);
      o2 = wmma_bf16(pah, vbh, o2); o2 = wmma_bf16(pal, vbh, o2); o2 = wmma_bf16(pah, vbl, o2);
    }
    {
      v16bf vbh = frag_ld(&sVh[48 + mrow], 64, half);
      v16bf vbl = frag_ld(&sVl[48 + mrow], 64, half);
      o3 = wmma_bf16(pah, vbh, o3); o3 = wmma_bf16(pal, vbh, o3); o3 = wmma_bf16(pah, vbl, o3);
    }
    __syncthreads();
  }
#pragma unroll
  for (int r = 0; r < 8; ++r) {
    int srow = qr + r + 8 * half;
    float inv = 1.0f / lrun[r];
    size_t base = ((size_t)(b * S_ + srow)) * D_ + h * DK_;
    AO[base + mrow]      = o0[r] * inv;
    AO[base + 16 + mrow] = o1[r] * inv;
    AO[base + 32 + mrow] = o2[r] * inv;
    AO[base + 48 + mrow] = o3[r] * inv;
  }
}

// ---------- mHC: per-(token,group) GS=4 block with 5-iter Sinkhorn ----------
__global__ void mhc_kernel(const float* __restrict__ x, const float* __restrict__ f,
                           const float* __restrict__ rms_w,
                           const float* __restrict__ phi_pre, const float* __restrict__ phi_post,
                           const float* __restrict__ phi_res,
                           const float* __restrict__ a_pre, const float* __restrict__ a_post,
                           const float* __restrict__ a_res,
                           const float* __restrict__ b_pre, const float* __restrict__ b_post,
                           const float* __restrict__ b_res,
                           float* __restrict__ out) {
  int idx = blockIdx.x * blockDim.x + threadIdx.x;
  if (idx >= B_ * S_ * G_) return;
  int g = idx & (G_ - 1);
  int bs = idx >> 8;
  size_t base = (size_t)bs * D_ + g * GS_;
  float xv[4], fv[4], ss = 0.f;
#pragma unroll
  for (int j = 0; j < 4; ++j) { xv[j] = x[base + j]; fv[j] = f[base + j]; ss += xv[j] * xv[j]; }
  float rinv = rsqrtf(ss * 0.25f + EPS_);
  float xn[4];
#pragma unroll
  for (int j = 0; j < 4; ++j) xn[j] = xv[j] * rms_w[j] * rinv;
  float hpre[4], hpost[4];
#pragma unroll
  for (int k = 0; k < 4; ++k) {
    float sp = 0.f, so = 0.f;
#pragma unroll
    for (int j = 0; j < 4; ++j) {
      sp += xn[j] * phi_pre[(g * 4 + j) * 4 + k];
      so += xn[j] * phi_post[(g * 4 + j) * 4 + k];
    }
    hpre[k]  = a_pre[g * 4 + k] * sp + b_pre[g * 4 + k];
    hpost[k] = a_post[g * 4 + k] * so + b_post[g * 4 + k];
  }
  float Mm[4][4];
#pragma unroll
  for (int i2 = 0; i2 < 4; ++i2)
#pragma unroll
    for (int j = 0; j < 4; ++j) {
      int k2 = i2 * 4 + j;
      float pr = 0.f;
#pragma unroll
      for (int jj = 0; jj < 4; ++jj) pr += xn[jj] * phi_res[(g * 4 + jj) * 16 + k2];
      float hres = a_res[(g * 4 + i2) * 4 + j] * pr + b_res[(g * 4 + i2) * 4 + j];
      Mm[i2][j] = __expf(hres);
    }
#pragma unroll
  for (int it = 0; it < 5; ++it) {
#pragma unroll
    for (int i2 = 0; i2 < 4; ++i2) {
      float inv = 1.f / (Mm[i2][0] + Mm[i2][1] + Mm[i2][2] + Mm[i2][3] + 1e-9f);
#pragma unroll
      for (int j = 0; j < 4; ++j) Mm[i2][j] *= inv;
    }
#pragma unroll
    for (int j = 0; j < 4; ++j) {
      float inv = 1.f / (Mm[0][j] + Mm[1][j] + Mm[2][j] + Mm[3][j] + 1e-9f);
#pragma unroll
      for (int i2 = 0; i2 < 4; ++i2) Mm[i2][j] *= inv;
    }
  }
  float gated[4];
#pragma unroll
  for (int j = 0; j < 4; ++j) gated[j] = xv[j] / (1.f + __expf(-hpre[j]));
#pragma unroll
  for (int i2 = 0; i2 < 4; ++i2) {
    float r = Mm[i2][0] * gated[0] + Mm[i2][1] * gated[1] + Mm[i2][2] * gated[2] + Mm[i2][3] * gated[3];
    float cr = 2.f * fv[i2] / (1.f + __expf(-hpost[i2]));
    out[base + i2] = r + cr;
  }
}

// ---------- driver ----------
extern "C" void kernel_launch(void* const* d_in, const int* in_sizes, int n_in,
                              void* d_out, int out_size, void* d_ws, size_t ws_size,
                              hipStream_t stream) {
  (void)in_sizes; (void)n_in; (void)out_size; (void)ws_size;
  const float* x       = (const float*)d_in[0];
  const float* ln1_w   = (const float*)d_in[1];
  const float* ln2_w   = (const float*)d_in[2];
  const float* wq      = (const float*)d_in[3];
  const float* wk      = (const float*)d_in[4];
  const float* wv      = (const float*)d_in[5];
  const float* wo      = (const float*)d_in[6];
  const float* w1      = (const float*)d_in[7];
  const float* w2      = (const float*)d_in[8];
  const float* w3      = (const float*)d_in[9];
  const float* rms_w   = (const float*)d_in[10];
  const float* phi_pre = (const float*)d_in[11];
  const float* phi_post= (const float*)d_in[12];
  const float* phi_res = (const float*)d_in[13];
  const float* a_pre   = (const float*)d_in[14];
  const float* a_post  = (const float*)d_in[15];
  const float* a_res   = (const float*)d_in[16];
  const float* b_pre   = (const float*)d_in[17];
  const float* b_post  = (const float*)d_in[18];
  const float* b_res   = (const float*)d_in[19];
  float* out = (float*)d_out;

  const size_t DDe  = (size_t)D_ * D_;
  const size_t DFDe = (size_t)DFF_ * D_;
  const size_t BSDe = (size_t)B_ * S_ * D_;
  const size_t BSFe = (size_t)B_ * S_ * DFF_;

  char* p = (char*)d_ws;
  auto alloc = [&](size_t bytes) -> char* {
    char* r = p; p += (bytes + 255) & ~(size_t)255; return r;
  };
  // bf16 hi/lo weight copies
  unsigned short* WQH = (unsigned short*)alloc(DDe * 2);
  unsigned short* WQL = (unsigned short*)alloc(DDe * 2);
  unsigned short* WKH = (unsigned short*)alloc(DDe * 2);
  unsigned short* WKL = (unsigned short*)alloc(DDe * 2);
  unsigned short* WVH = (unsigned short*)alloc(DDe * 2);
  unsigned short* WVL = (unsigned short*)alloc(DDe * 2);
  unsigned short* WOH = (unsigned short*)alloc(DDe * 2);
  unsigned short* WOL = (unsigned short*)alloc(DDe * 2);
  unsigned short* W1H = (unsigned short*)alloc(DFDe * 2);
  unsigned short* W1L = (unsigned short*)alloc(DFDe * 2);
  unsigned short* W3H = (unsigned short*)alloc(DFDe * 2);
  unsigned short* W3L = (unsigned short*)alloc(DFDe * 2);
  unsigned short* W2H = (unsigned short*)alloc(DFDe * 2);
  unsigned short* W2L = (unsigned short*)alloc(DFDe * 2);
  // activations
  unsigned short* XNH = (unsigned short*)alloc(BSDe * 2);
  unsigned short* XNL = (unsigned short*)alloc(BSDe * 2);
  unsigned short* QHb = (unsigned short*)alloc(BSDe * 2);
  unsigned short* QLb = (unsigned short*)alloc(BSDe * 2);
  unsigned short* KHb = (unsigned short*)alloc(BSDe * 2);
  unsigned short* KLb = (unsigned short*)alloc(BSDe * 2);
  unsigned short* VHb = (unsigned short*)alloc(BSDe * 2);
  unsigned short* VLb = (unsigned short*)alloc(BSDe * 2);
  unsigned short* GHb = (unsigned short*)alloc(BSFe * 2);
  unsigned short* GLb = (unsigned short*)alloc(BSFe * 2);
  float* QF = (float*)alloc(BSDe * 4);
  float* KF = (float*)alloc(BSDe * 4);
  float* VF = (float*)alloc(BSDe * 4);
  float* AO = (float*)alloc(BSDe * 4);
  float* X1 = (float*)alloc(BSDe * 4);
  float* L1 = (float*)alloc(BSFe * 4);
  float* L3 = (float*)alloc(BSFe * 4);
  float* YF = (float*)alloc(BSDe * 4);

  // 1) split weights to bf16 hi/lo
  pack_f32_bf16hl<<<(DDe + 255) / 256, 256, 0, stream>>>(wq, WQH, WQL, (int)DDe);
  pack_f32_bf16hl<<<(DDe + 255) / 256, 256, 0, stream>>>(wk, WKH, WKL, (int)DDe);
  pack_f32_bf16hl<<<(DDe + 255) / 256, 256, 0, stream>>>(wv, WVH, WVL, (int)DDe);
  pack_f32_bf16hl<<<(DDe + 255) / 256, 256, 0, stream>>>(wo, WOH, WOL, (int)DDe);
  pack_f32_bf16hl<<<(DFDe + 255) / 256, 256, 0, stream>>>(w1, W1H, W1L, (int)DFDe);
  pack_f32_bf16hl<<<(DFDe + 255) / 256, 256, 0, stream>>>(w3, W3H, W3L, (int)DFDe);
  pack_f32_bf16hl<<<(DFDe + 255) / 256, 256, 0, stream>>>(w2, W2H, W2L, (int)DFDe);

  const dim3 gq(D_ / 64, (B_ * S_) / 64);
  const dim3 gf(DFF_ / 64, (B_ * S_) / 64);

  // 2) attention path
  rmsnorm_pack<<<B_ * S_, 256, 0, stream>>>(x, ln1_w, XNH, XNL);
  gemm_bf16x3<<<gq, 256, 0, stream>>>(XNH, XNL, WQH, WQL, nullptr, QF, D_, D_);
  gemm_bf16x3<<<gq, 256, 0, stream>>>(XNH, XNL, WKH, WKL, nullptr, KF, D_, D_);
  gemm_bf16x3<<<gq, 256, 0, stream>>>(XNH, XNL, WVH, WVL, nullptr, VF, D_, D_);
  rope_pack<<<(B_ * S_ * H_ * (DK_ / 2)) / 256, 256, 0, stream>>>(QF, KF, QHb, QLb, KHb, KLb);
  pack_f32_bf16hl<<<(BSDe + 255) / 256, 256, 0, stream>>>(VF, VHb, VLb, (int)BSDe);
  attn_flash<<<dim3(S_ / 64, H_, B_), 128, 0, stream>>>(QHb, QLb, KHb, KLb, VHb, VLb, AO);
  pack_f32_bf16hl<<<(BSDe + 255) / 256, 256, 0, stream>>>(AO, QHb, QLb, (int)BSDe);  // reuse Q bufs
  gemm_bf16x3<<<gq, 256, 0, stream>>>(QHb, QLb, WOH, WOL, x, X1, D_, D_);            // x1 = x + attn@wo^T

  // 3) FFN path
  rmsnorm_pack<<<B_ * S_, 256, 0, stream>>>(X1, ln2_w, XNH, XNL);
  gemm_bf16x3<<<gf, 256, 0, stream>>>(XNH, XNL, W1H, W1L, nullptr, L1, DFF_, D_);
  gemm_bf16x3<<<gf, 256, 0, stream>>>(XNH, XNL, W3H, W3L, nullptr, L3, DFF_, D_);
  swiglu_pack<<<(BSFe + 255) / 256, 256, 0, stream>>>(L1, L3, GHb, GLb, (int)BSFe);
  gemm_bf16x3<<<gq, 256, 0, stream>>>(GHb, GLb, W2H, W2L, X1, YF, D_, DFF_);         // y = x1 + glu@w2^T

  // 4) mHC head -> output
  mhc_kernel<<<(B_ * S_ * G_) / 256, 256, 0, stream>>>(X1, YF, rms_w, phi_pre, phi_post, phi_res,
                                                       a_pre, a_post, a_res, b_pre, b_post, b_res, out);
}
